// ConformerBlock_39316130627955
// MI455X (gfx1250) — matrix operand order repaired and checked by reading, not compiled, not run on hardware
//
#include <hip/hip_runtime.h>
#include <hip/hip_bf16.h>
#include <math.h>

// ---------------- problem dims ----------------
#define Bb   4
#define Ts   1024
#define Dm   512
#define Hh   8
#define DKc  64
#define DFFc 2048
#define KC   31
#define BT   (Bb*Ts)
#define PEM  2176      // (2T-1)=2047 rows padded up to a multiple of 128; rows >=2047 are zero

typedef __attribute__((ext_vector_type(16))) __bf16 bf16x16;
typedef __attribute__((ext_vector_type(8)))  float  f32x8;

union FragH { bf16x16 v; unsigned int u[8]; unsigned short u16[16]; };

__device__ __forceinline__ float sigf(float x) { return 1.f / (1.f + __expf(-x)); }

// ---------------- CDNA5 feature detection (compile-safe probes) ----------------
#define AS3 __attribute__((address_space(3)))
#define LDS_OFS(p) ((unsigned int)(size_t)(AS3 const void*)(const void*)(p))

#if __has_builtin(__builtin_amdgcn_tensor_load_to_lds) && __has_builtin(__builtin_amdgcn_s_wait_tensorcnt)
#define USE_TDM 1
#else
#define USE_TDM 0
#endif

#if __has_builtin(__builtin_amdgcn_global_load_async_to_lds_b128) && __has_builtin(__builtin_amdgcn_s_wait_asynccnt)
#define USE_ASYNC 1
// the builtin takes generic pointers to 16-byte int vectors (per clang diagnostic)
typedef int v4i_vs __attribute__((__vector_size__(16)));
#else
#define USE_ASYNC 0
#endif

#if USE_TDM
typedef __attribute__((ext_vector_type(4))) unsigned int u32x4;
typedef __attribute__((ext_vector_type(8))) int          i32x8;
typedef __attribute__((ext_vector_type(4))) int          i32x4;

// Issue a 2D TDM load: global (row-major, stride0 elements of 2B) -> LDS (contiguous tile).
// D# packing per CDNA5 ISA ch.8: group0 = {count/type, lds_addr, global_addr}, group1 = dims.
__device__ __forceinline__ void tdm_load_2d(const void* gptr, unsigned int lds_off,
                                            unsigned int tile_d0, unsigned int tile_d1,
                                            unsigned int tens_d0, unsigned int tens_d1,
                                            unsigned long stride0_elems) {
    unsigned long ga = (unsigned long)gptr;
    u32x4 g0;
    g0.x = 1u;                                                  // count=1, user descriptor
    g0.y = lds_off;                                             // lds_addr (bytes)
    g0.z = (unsigned int)ga;                                    // global_addr[31:0]
    g0.w = (unsigned int)((ga >> 32) & 0x01FFFFFFu) | (2u << 30); // addr[56:32] | type=2
    i32x8 g1;
    g1[0] = (int)(1u << 16);                                    // data_size=2B; mask/pad/iter=0
    g1[1] = (int)((tens_d0 & 0xFFFFu) << 16);                   // tensor_dim0[15:0] @ bits 63:48
    g1[2] = (int)(((tens_d0 >> 16) & 0xFFFFu) | ((tens_d1 & 0xFFFFu) << 16)); // dim0 hi | dim1 lo
    g1[3] = (int)(((tens_d1 >> 16) & 0xFFFFu) | ((tile_d0 & 0xFFFFu) << 16)); // dim1 hi | tile_dim0
    g1[4] = (int)(tile_d1 & 0xFFFFu);                           // tile_dim1; tile_dim2=0
    g1[5] = (int)(unsigned int)(stride0_elems & 0xFFFFFFFFu);   // tensor_dim0_stride[31:0]
    g1[6] = (int)(unsigned int)((stride0_elems >> 32) & 0xFFFFu);
    g1[7] = 0;
    i32x4 z4 = {0, 0, 0, 0};
#if __clang_major__ >= 23
    i32x8 z8 = {0, 0, 0, 0, 0, 0, 0, 0};
    __builtin_amdgcn_tensor_load_to_lds(g0, g1, z4, z4, z8, 0);
#else
    __builtin_amdgcn_tensor_load_to_lds(g0, g1, z4, z4, 0);
#endif
}
#endif  // USE_TDM

// ---------------- f32 -> bf16 convert ----------------
__global__ void cvt_f32_to_bf16(const float* __restrict__ in, __bf16* __restrict__ out, int n) {
    int i = blockIdx.x * blockDim.x + threadIdx.x;
    if (i < n) out[i] = (__bf16)in[i];
}

// ---------------- LayerNorm over rows of Dm ----------------
__global__ __launch_bounds__(256) void ln_kernel(const float* __restrict__ x,
                                                 const float* __restrict__ g,
                                                 const float* __restrict__ bta,
                                                 float* __restrict__ outF,
                                                 __bf16* __restrict__ outH) {
    __shared__ float sbuf[256];
    int row = blockIdx.x, tid = threadIdx.x;
    const float* xr = x + (size_t)row * Dm;
    float v0 = xr[tid], v1 = xr[tid + 256];
    sbuf[tid] = v0 + v1;
    __syncthreads();
    for (int o = 128; o > 0; o >>= 1) { if (tid < o) sbuf[tid] += sbuf[tid + o]; __syncthreads(); }
    float mean = sbuf[0] * (1.f / Dm);
    __syncthreads();
    float d0 = v0 - mean, d1 = v1 - mean;
    sbuf[tid] = d0 * d0 + d1 * d1;
    __syncthreads();
    for (int o = 128; o > 0; o >>= 1) { if (tid < o) sbuf[tid] += sbuf[tid + o]; __syncthreads(); }
    float r = rsqrtf(sbuf[0] * (1.f / Dm) + 1e-5f);
    float y0 = d0 * r * g[tid]       + bta[tid];
    float y1 = d1 * r * g[tid + 256] + bta[tid + 256];
    size_t base = (size_t)row * Dm;
    if (outF) { outF[base + tid] = y0; outF[base + tid + 256] = y1; }
    if (outH) { outH[base + tid] = (__bf16)y0; outH[base + tid + 256] = (__bf16)y1; }
}

// ---------------- sinusoidal relative position embedding (bf16, padded rows zero) ----------------
__global__ void pe_kernel(__bf16* __restrict__ pe) {
    int i = blockIdx.x * blockDim.x + threadIdx.x;    // over PEM*Dm
    int d = i % Dm, p = i / Dm;
    float val = 0.f;
    if (p < 2 * Ts - 1) {
        float pos  = (float)(p - (Ts - 1));
        float freq = __expf(-(float)(d & ~1) * (logf(10000.f) / Dm));
        float ang  = pos * freq;
        val = (d & 1) ? __cosf(ang) : __sinf(ang);
    }
    pe[i] = (__bf16)val;
}

// ---------------- qu = q + pos_u[h], qv = q + pos_v[h] (bf16) ----------------
__global__ void quqv_kernel(const float* __restrict__ q, const float* __restrict__ pu,
                            const float* __restrict__ pv, __bf16* __restrict__ qu,
                            __bf16* __restrict__ qv, int n) {
    int i = blockIdx.x * blockDim.x + threadIdx.x;
    if (i >= n) return;
    int d = i % Dm;                 // pos_u is [H,DK] flat == [Dm]
    float qq = q[i];
    qu[i] = (__bf16)(qq + pu[d]);
    qv[i] = (__bf16)(qq + pv[d]);
}

// ---------------- WMMA bf16 GEMM: out = act(A@B + bias); optional out = resid + alpha*out ------
// A[M,K] bf16 row-major, B[K,N] bf16 row-major, M%128==0, N%128==0, K%32==0.
// Block tile 128x128, 8 waves (2x4), wave tile 64x32, K-step 32.
// A-tile (and B-tile row-major staging) pulled by the Tensor Data Mover when available.
__global__ __launch_bounds__(256) void gemm_bf16(const __bf16* __restrict__ A,
                                                 const __bf16* __restrict__ Bw,
                                                 const float* __restrict__ bias,
                                                 const float* __restrict__ resid,
                                                 float alpha,
                                                 float* __restrict__ outF,
                                                 __bf16* __restrict__ outH,
                                                 int M, int N, int K, int act) {
    __shared__ unsigned int   lA[128 * 16];   // A tile: 128 rows x 32 bf16 (16 dwords/row)
    __shared__ unsigned short lB[128 * 32];   // B tile transposed: [n][k]
#if USE_TDM
    __shared__ unsigned short lBrm[32 * 128]; // B tile row-major staging for TDM
#endif
    int tid  = threadIdx.x;
    int lane = tid & 31, wid = tid >> 5;
    int wm   = wid & 1,  wn  = wid >> 1;
    int lm   = lane & 15, hf = lane >> 4;
    int m0   = blockIdx.y * 128, n0 = blockIdx.x * 128;
    (void)M;

    f32x8 acc[4][2];
#pragma unroll
    for (int i = 0; i < 4; i++)
#pragma unroll
        for (int j = 0; j < 2; j++)
#pragma unroll
            for (int r = 0; r < 8; r++) acc[i][j][r] = 0.f;

    for (int k0 = 0; k0 < K; k0 += 32) {
        __syncthreads();
#if USE_TDM
        if (wid == 0) {     // one wave drives the DMA; EXEC ignored by tensor ops
            tdm_load_2d(A + (size_t)m0 * K + k0, LDS_OFS(lA),
                        /*tile*/ 32, 128, /*tensor*/ (unsigned)K, (unsigned)M, (unsigned long)K);
            tdm_load_2d(Bw + (size_t)k0 * N + n0, LDS_OFS(lBrm),
                        /*tile*/ 128, 32, /*tensor*/ (unsigned)N, (unsigned)K, (unsigned long)N);
            __builtin_amdgcn_s_wait_tensorcnt(0);
        }
        __syncthreads();
        // transpose lBrm[k][n] -> lB[n][k]
#pragma unroll
        for (int r = 0; r < 8; r++) {
            int li = tid + r * 256;            // 0..2047 dwords
            int kr = li >> 6, cu = li & 63;
            unsigned int w = ((const unsigned int*)lBrm)[kr * 64 + cu];
            int n = cu * 2;
            lB[n * 32 + kr]       = (unsigned short)(w & 0xffffu);
            lB[(n + 1) * 32 + kr] = (unsigned short)(w >> 16);
        }
#else
        // stage A: 1024 x uint2
#pragma unroll
        for (int r = 0; r < 4; r++) {
            int li = tid + r * 256;           // 0..1023
            int row = li >> 3, c2 = li & 7;   // 8 uint2 per row
            uint2 w = *(const uint2*)(A + (size_t)(m0 + row) * K + k0 + c2 * 4);
            ((uint2*)lA)[li] = w;
        }
        // stage B transposed: 2048 x uint
#pragma unroll
        for (int r = 0; r < 8; r++) {
            int li = tid + r * 256;           // 0..2047
            int kr = li >> 6, cu = li & 63;   // 64 dwords per k-row
            unsigned int w = *(const unsigned int*)(Bw + (size_t)(k0 + kr) * N + n0 + cu * 2);
            int n = cu * 2;
            lB[n * 32 + kr]       = (unsigned short)(w & 0xffffu);
            lB[(n + 1) * 32 + kr] = (unsigned short)(w >> 16);
        }
#endif
        __syncthreads();
        if (k0 + 32 < K) {  // prefetch next tiles (global_prefetch_b8)
            __builtin_prefetch(A + (size_t)(m0 + (tid >> 1)) * K + (k0 + 32), 0, 1);
            __builtin_prefetch(Bw + (size_t)(k0 + 32 + (tid & 31)) * N + n0, 0, 1);
        }

        FragH fa[4], fb[2];
#pragma unroll
        for (int ms = 0; ms < 4; ms++) {
            int rowa = wm * 64 + ms * 16 + lm;
#pragma unroll
            for (int v = 0; v < 8; v++) {
                int u = hf * 8 + v;
                fa[ms].u[v] = lA[rowa * 16 + u];
            }
        }
        const unsigned int* lBu = (const unsigned int*)lB;
#pragma unroll
        for (int ns = 0; ns < 2; ns++) {
            int n = wn * 32 + ns * 16 + lm;
#pragma unroll
            for (int v = 0; v < 8; v++) fb[ns].u[v] = lBu[n * 16 + hf * 8 + v];
        }
#pragma unroll
        for (int ms = 0; ms < 4; ms++)
#pragma unroll
            for (int ns = 0; ns < 2; ns++)
                acc[ms][ns] = __builtin_amdgcn_wmma_f32_16x16x32_bf16(
                    false, fa[ms].v, false, fb[ns].v, (short)0, acc[ms][ns], false, false);
    }

#pragma unroll
    for (int ms = 0; ms < 4; ms++)
#pragma unroll
        for (int ns = 0; ns < 2; ns++) {
            int col = n0 + wn * 32 + ns * 16 + lm;
            float bv = bias ? bias[col] : 0.f;
#pragma unroll
            for (int r = 0; r < 8; r++) {
                int row = m0 + wm * 64 + ms * 16 + r + 8 * hf;
                float v = acc[ms][ns][r] + bv;
                if (act == 1) v = v * sigf(v);          // SiLU
                size_t idx = (size_t)row * N + col;
                if (resid) v = resid[idx] + alpha * v;
                if (outF) outF[idx] = v;
                if (outH) outH[idx] = (__bf16)v;
            }
        }
}

// ---------------- fused flash attention with TXL relative position -------------------
// grid = B*H*(T/64), 128 threads (4 waves); each wave owns a 16-row t-tile.
// score(t,s) = (qu[t]·k[s] + qv[t]·pp[s-t+T]) / sqrt(DK); streaming softmax; O = P@V.
// K/V tiles staged to LDS with async global->LDS DMA when available.
__global__ __launch_bounds__(128) void attn_kernel(const __bf16* __restrict__ qu,
                                                   const __bf16* __restrict__ qv,
                                                   const __bf16* __restrict__ kmat,
                                                   const __bf16* __restrict__ vmat,
                                                   const __bf16* __restrict__ pp,
                                                   __bf16* __restrict__ o) {
    __shared__ __bf16 lK[32 * 64];            // k tile row-major [s][dk]
    __shared__ __bf16 lV[32 * 64];            // v tile row-major [s][dk]
    __shared__ __bf16 lP[4][16 * 32];         // per-wave probability tile
    int tid  = threadIdx.x;
    int lane = tid & 31, wid = tid >> 5;
    int lm   = lane & 15, hf = lane >> 4;
    int tb = blockIdx.x % (Ts / 64);
    int h  = (blockIdx.x / (Ts / 64)) % Hh;
    int b  = blockIdx.x / ((Ts / 64) * Hh);
    int t0 = tb * 64 + wid * 16;

    const unsigned int* quU = (const unsigned int*)qu;
    const unsigned int* qvU = (const unsigned int*)qv;
    const unsigned int* ppU = (const unsigned int*)pp;

    // A-frags for qu/qv: 16 rows x 64 (two K=32 slices)
    FragH fqu[2], fqv[2];
    int trow = t0 + lm;
#pragma unroll
    for (int ks = 0; ks < 2; ks++)
#pragma unroll
        for (int v = 0; v < 8; v++) {
            int kk = ks * 32 + hf * 16 + v * 2;
            size_t e = (((size_t)b * Ts + trow) * Dm + h * DKc + kk) >> 1;
            fqu[ks].u[v] = quU[e];
            fqv[ks].u[v] = qvU[e];
        }

    f32x8 oacc[4];
#pragma unroll
    for (int f = 0; f < 4; f++)
#pragma unroll
        for (int r = 0; r < 8; r++) oacc[f][r] = 0.f;
    float mrun[8], lrun[8];
#pragma unroll
    for (int r = 0; r < 8; r++) { mrun[r] = -1e30f; lrun[r] = 0.f; }
    const float rscale = 0.125f;   // 1/sqrt(64)

    for (int s0 = 0; s0 < Ts; s0 += 32) {
        __syncthreads();
#if USE_ASYNC
        // async DMA: each thread copies two 16B chunks of each tile straight into LDS
#pragma unroll
        for (int r = 0; r < 2; r++) {
            int li = tid + r * 128;            // 0..255 : 32 rows x 8 chunks
            int sr = li >> 3, c = li & 7;
            const __bf16* gk = kmat + ((size_t)b * Ts + s0 + sr) * Dm + h * DKc + c * 8;
            const __bf16* gv = vmat + ((size_t)b * Ts + s0 + sr) * Dm + h * DKc + c * 8;
            __builtin_amdgcn_global_load_async_to_lds_b128(
                (v4i_vs*)gk, (v4i_vs*)&lK[sr * 64 + c * 8], 0, 0);
            __builtin_amdgcn_global_load_async_to_lds_b128(
                (v4i_vs*)gv, (v4i_vs*)&lV[sr * 64 + c * 8], 0, 0);
        }
        __builtin_amdgcn_s_wait_asynccnt(0);
#else
#pragma unroll
        for (int r = 0; r < 4; r++) {
            int li = tid + r * 128;            // 0..511 uint2 per tile
            int sr = li >> 4, c2 = li & 15;    // 16 uint2 per row
            size_t goff = ((size_t)b * Ts + s0 + sr) * Dm + h * DKc + c2 * 4;
            ((uint2*)lK)[sr * 16 + c2] = *(const uint2*)(kmat + goff);
            ((uint2*)lV)[sr * 16 + c2] = *(const uint2*)(vmat + goff);
        }
#endif
        __syncthreads();

        const unsigned int* lKu = (const unsigned int*)lK;
        float sv[2][8];
#pragma unroll
        for (int sub = 0; sub < 2; sub++) {
            int ss  = s0 + sub * 16;
            int p00 = Ts + ss - t0;               // rel-shift base: p = s - t + T
            f32x8 ac, d1, d2;
#pragma unroll
            for (int r = 0; r < 8; r++) { ac[r] = 0.f; d1[r] = 0.f; d2[r] = 0.f; }
#pragma unroll
            for (int ks = 0; ks < 2; ks++) {
                FragH fk, f1, f2;
                int srow = sub * 16 + lm;          // row inside the 32-row LDS tile
#pragma unroll
                for (int v = 0; v < 8; v++) {
                    int kk = ks * 32 + hf * 16 + v * 2;
                    fk.u[v] = lKu[srow * 32 + (kk >> 1)];
                    f1.u[v] = ppU[(((size_t)(p00 - 15 + lm)) * Dm + h * DKc + kk) >> 1];
                    f2.u[v] = ppU[(((size_t)(p00 +  1 + lm)) * Dm + h * DKc + kk) >> 1];
                }
                ac = __builtin_amdgcn_wmma_f32_16x16x32_bf16(false, fqu[ks].v, false, fk.v, (short)0, ac, false, false);
                d1 = __builtin_amdgcn_wmma_f32_16x16x32_bf16(false, fqv[ks].v, false, f1.v, (short)0, d1, false, false);
                d2 = __builtin_amdgcn_wmma_f32_16x16x32_bf16(false, fqv[ks].v, false, f2.v, (short)0, d2, false, false);
            }
            // rel-shift gather: bd(i,j) lives in lane (j-i+15)&15 of D1 (j<=i) or D2 (j>i)
#pragma unroll
            for (int r = 0; r < 8; r++) {
                int irow = r + 8 * hf;
                int idx  = lm - irow + 15;        // 0..30
                int src  = (hf << 4) | (idx & 15);
                float v1 = __shfl(d1[r], src, 32);
                float v2 = __shfl(d2[r], src, 32);
                float bd = (idx < 16) ? v1 : v2;
                sv[sub][r] = (ac[r] + bd) * rscale;
            }
        }

        // streaming softmax update over the 32 columns
#pragma unroll
        for (int r = 0; r < 8; r++) {
            float mv = fmaxf(sv[0][r], sv[1][r]);
#pragma unroll
            for (int off = 8; off > 0; off >>= 1) mv = fmaxf(mv, __shfl_xor(mv, off, 16));
            float mnew = fmaxf(mrun[r], mv);
            float corr = __expf(mrun[r] - mnew);
            float p0 = __expf(sv[0][r] - mnew);
            float p1 = __expf(sv[1][r] - mnew);
            float ps = p0 + p1;
#pragma unroll
            for (int off = 8; off > 0; off >>= 1) ps += __shfl_xor(ps, off, 16);
            lrun[r] = lrun[r] * corr + ps;
            mrun[r] = mnew;
#pragma unroll
            for (int f = 0; f < 4; f++) oacc[f][r] *= corr;
            int prow = r + 8 * hf;
            lP[wid][prow * 32 + lm]      = (__bf16)p0;
            lP[wid][prow * 32 + 16 + lm] = (__bf16)p1;
        }

        // O += P @ V    (P: 16x32 A-frag from LDS; V: 32x64 -> 4 B-frags from row-major LDS)
        FragH fp;
        const unsigned int* lPu = (const unsigned int*)&lP[wid][0];
#pragma unroll
        for (int v = 0; v < 8; v++) {
            int kk = hf * 16 + v * 2;
            fp.u[v] = lPu[lm * 16 + (kk >> 1)];
        }
#pragma unroll
        for (int f = 0; f < 4; f++) {
            FragH fv;
            int n = f * 16 + lm;
#pragma unroll
            for (int v = 0; v < 8; v++) {
                int kk = hf * 16 + v * 2;
                fv.u16[v * 2]     = *(const unsigned short*)&lV[kk * 64 + n];
                fv.u16[v * 2 + 1] = *(const unsigned short*)&lV[(kk + 1) * 64 + n];
            }
            oacc[f] = __builtin_amdgcn_wmma_f32_16x16x32_bf16(false, fp.v, false, fv.v, (short)0, oacc[f], false, false);
        }
    }

#pragma unroll
    for (int r = 0; r < 8; r++) {
        float inv = 1.f / lrun[r];
        int to = t0 + r + 8 * hf;
#pragma unroll
        for (int f = 0; f < 4; f++)
            o[((size_t)b * Ts + to) * Dm + h * DKc + f * 16 + lm] = (__bf16)(oacc[f][r] * inv);
    }
}

// ---------------- GLU ----------------
__global__ void glu_kernel(const float* __restrict__ x, float* __restrict__ out, int n) {
    int i = blockIdx.x * blockDim.x + threadIdx.x;
    if (i >= n) return;
    int r = i / Dm, c = i % Dm;
    float a = x[(size_t)r * (2 * Dm) + c];
    float g = x[(size_t)r * (2 * Dm) + Dm + c];
    out[i] = a * sigf(g);
}

// ---------------- depthwise conv along T (channels-last, zero pad) ----------------
__global__ void dwconv_kernel(const float* __restrict__ x, const float* __restrict__ w,
                              const float* __restrict__ bias, float* __restrict__ out) {
    int i = blockIdx.x * blockDim.x + threadIdx.x;   // B*T*D
    int d = i % Dm, t = (i / Dm) % Ts, b = i / (Dm * Ts);
    float acc = bias[d];
#pragma unroll
    for (int k = 0; k < KC; k++) {
        int tt = t + k - (KC - 1) / 2;
        if (tt >= 0 && tt < Ts) acc += x[((size_t)b * Ts + tt) * Dm + d] * w[d * KC + k];
    }
    out[i] = acc;
}

// ---------------- BatchNorm stats per channel (mean, var) ----------------
__global__ __launch_bounds__(256) void bn_stats_kernel(const float* __restrict__ x,
                                                       float* __restrict__ stats) {
    __shared__ float s1[256], s2[256];
    int d = blockIdx.x, tid = threadIdx.x;
    float a = 0.f, b2 = 0.f;
    for (int i = tid; i < BT; i += 256) {
        float v = x[(size_t)i * Dm + d];
        a += v; b2 += v * v;
    }
    s1[tid] = a; s2[tid] = b2;
    __syncthreads();
    for (int o = 128; o > 0; o >>= 1) {
        if (tid < o) { s1[tid] += s1[tid + o]; s2[tid] += s2[tid + o]; }
        __syncthreads();
    }
    if (tid == 0) {
        float m = s1[0] * (1.f / BT);
        stats[d] = m;
        stats[Dm + d] = s2[0] * (1.f / BT) - m * m;
    }
}

// ---------------- BN apply + SiLU -> bf16 ----------------
__global__ void bn_apply_kernel(const float* __restrict__ x, const float* __restrict__ stats,
                                const float* __restrict__ g, const float* __restrict__ bta,
                                __bf16* __restrict__ out, int n) {
    int i = blockIdx.x * blockDim.x + threadIdx.x;
    if (i >= n) return;
    int d = i % Dm;
    float v = (x[i] - stats[d]) * rsqrtf(stats[Dm + d] + 1e-5f) * g[d] + bta[d];
    v = v * sigf(v);
    out[i] = (__bf16)v;
}

// ================= host orchestration =================
extern "C" void kernel_launch(void* const* d_in, const int* in_sizes, int n_in,
                              void* d_out, int out_size, void* d_ws, size_t ws_size,
                              hipStream_t stream) {
    (void)in_sizes; (void)n_in; (void)out_size; (void)ws_size;
    const float* x       = (const float*)d_in[0];
    const float* f1_lng  = (const float*)d_in[1];
    const float* f1_lnb  = (const float*)d_in[2];
    const float* f1_w1   = (const float*)d_in[3];
    const float* f1_b1   = (const float*)d_in[4];
    const float* f1_w2   = (const float*)d_in[5];
    const float* f1_b2   = (const float*)d_in[6];
    const float* f2_lng  = (const float*)d_in[7];
    const float* f2_lnb  = (const float*)d_in[8];
    const float* f2_w1   = (const float*)d_in[9];
    const float* f2_b1   = (const float*)d_in[10];
    const float* f2_w2   = (const float*)d_in[11];
    const float* f2_b2   = (const float*)d_in[12];
    const float* att_lng = (const float*)d_in[13];
    const float* att_lnb = (const float*)d_in[14];
    const float* wq      = (const float*)d_in[15];
    const float* wk      = (const float*)d_in[16];
    const float* wv      = (const float*)d_in[17];
    const float* wo      = (const float*)d_in[18];
    const float* bq      = (const float*)d_in[19];
    const float* bk      = (const float*)d_in[20];
    const float* bv      = (const float*)d_in[21];
    const float* bo      = (const float*)d_in[22];
    const float* wpos    = (const float*)d_in[23];
    const float* pos_u   = (const float*)d_in[24];
    const float* pos_v   = (const float*)d_in[25];
    const float* cv_lng  = (const float*)d_in[26];
    const float* cv_lnb  = (const float*)d_in[27];
    const float* pw1_w   = (const float*)d_in[28];
    const float* pw1_b   = (const float*)d_in[29];
    const float* dw_w    = (const float*)d_in[30];
    const float* dw_b    = (const float*)d_in[31];
    const float* bn_g    = (const float*)d_in[32];
    const float* bn_b    = (const float*)d_in[33];
    const float* pw2_w   = (const float*)d_in[34];
    const float* pw2_b   = (const float*)d_in[35];
    const float* out_lng = (const float*)d_in[36];
    const float* out_lnb = (const float*)d_in[37];

    // bump allocator over d_ws (~135 MB total)
    char* base = (char*)d_ws;
    size_t off = 0;
    auto alloc = [&](size_t bytes) -> void* {
        void* p = base + off;
        off += (bytes + 255) & ~(size_t)255;
        return p;
    };

    __bf16* wbf1w1 = (__bf16*)alloc((size_t)Dm * DFFc * 2);
    __bf16* wbf1w2 = (__bf16*)alloc((size_t)DFFc * Dm * 2);
    __bf16* wbf2w1 = (__bf16*)alloc((size_t)Dm * DFFc * 2);
    __bf16* wbf2w2 = (__bf16*)alloc((size_t)DFFc * Dm * 2);
    __bf16* wbq    = (__bf16*)alloc((size_t)Dm * Dm * 2);
    __bf16* wbk    = (__bf16*)alloc((size_t)Dm * Dm * 2);
    __bf16* wbv    = (__bf16*)alloc((size_t)Dm * Dm * 2);
    __bf16* wbo    = (__bf16*)alloc((size_t)Dm * Dm * 2);
    __bf16* wbpos  = (__bf16*)alloc((size_t)Dm * Dm * 2);
    __bf16* wbpw1  = (__bf16*)alloc((size_t)Dm * 2 * Dm * 2);
    __bf16* wbpw2  = (__bf16*)alloc((size_t)Dm * Dm * 2);
    __bf16* peb    = (__bf16*)alloc((size_t)PEM * Dm * 2);
    __bf16* ppb    = (__bf16*)alloc((size_t)PEM * Dm * 2);
    __bf16* lnb    = (__bf16*)alloc((size_t)BT * Dm * 2);    // reused by every LN stage
    __bf16* kb     = (__bf16*)alloc((size_t)BT * Dm * 2);
    __bf16* vb     = (__bf16*)alloc((size_t)BT * Dm * 2);
    __bf16* qub    = (__bf16*)alloc((size_t)BT * Dm * 2);
    __bf16* qvb    = (__bf16*)alloc((size_t)BT * Dm * 2);
    __bf16* ob     = (__bf16*)alloc((size_t)BT * Dm * 2);
    __bf16* h1b    = (__bf16*)alloc((size_t)BT * DFFc * 2);  // FFN hidden (reused f1/f2)
    __bf16* bnb    = (__bf16*)alloc((size_t)BT * Dm * 2);
    float*  qf     = (float*)alloc((size_t)BT * Dm * 4);
    float*  x1     = (float*)alloc((size_t)BT * Dm * 4);
    float*  x2     = (float*)alloc((size_t)BT * Dm * 4);
    float*  x3     = (float*)alloc((size_t)BT * Dm * 4);
    float*  x4     = (float*)alloc((size_t)BT * Dm * 4);
    float*  pw1f   = (float*)alloc((size_t)BT * 2 * Dm * 4);
    float*  gluf   = (float*)alloc((size_t)BT * Dm * 4);
    float*  convf  = (float*)alloc((size_t)BT * Dm * 4);
    float*  stats  = (float*)alloc((size_t)2 * Dm * 4);

    auto cvt = [&](const float* src, __bf16* dst, int n) {
        cvt_f32_to_bf16<<<(n + 255) / 256, 256, 0, stream>>>(src, dst, n);
    };
    auto gemm = [&](const __bf16* A, const __bf16* Bw, const float* bias, const float* resid,
                    float alpha, float* outF, __bf16* outH, int M, int N, int K, int act) {
        dim3 g(N / 128, M / 128);
        gemm_bf16<<<g, 256, 0, stream>>>(A, Bw, bias, resid, alpha, outF, outH, M, N, K, act);
    };

    // weight conversion f32 -> bf16
    cvt(f1_w1, wbf1w1, Dm * DFFc);
    cvt(f1_w2, wbf1w2, DFFc * Dm);
    cvt(f2_w1, wbf2w1, Dm * DFFc);
    cvt(f2_w2, wbf2w2, DFFc * Dm);
    cvt(wq,    wbq,    Dm * Dm);
    cvt(wk,    wbk,    Dm * Dm);
    cvt(wv,    wbv,    Dm * Dm);
    cvt(wo,    wbo,    Dm * Dm);
    cvt(wpos,  wbpos,  Dm * Dm);
    cvt(pw1_w, wbpw1,  Dm * 2 * Dm);
    cvt(pw2_w, wbpw2,  Dm * Dm);

    // ---- macaron FFN #1: x1 = x + 0.5 * W2(silu(W1 ln(x))) ----
    ln_kernel<<<BT, 256, 0, stream>>>(x, f1_lng, f1_lnb, nullptr, lnb);
    gemm(lnb, wbf1w1, f1_b1, nullptr, 1.f, nullptr, h1b, BT, DFFc, Dm, 1);
    gemm(h1b, wbf1w2, f1_b2, x, 0.5f, x1, nullptr, BT, Dm, DFFc, 0);

    // ---- attention: x2 = x1 + Wo(attn) ----
    ln_kernel<<<BT, 256, 0, stream>>>(x1, att_lng, att_lnb, nullptr, lnb);
    gemm(lnb, wbq, bq, nullptr, 1.f, qf, nullptr, BT, Dm, Dm, 0);
    gemm(lnb, wbk, bk, nullptr, 1.f, nullptr, kb, BT, Dm, Dm, 0);
    gemm(lnb, wbv, bv, nullptr, 1.f, nullptr, vb, BT, Dm, Dm, 0);
    quqv_kernel<<<(BT * Dm + 255) / 256, 256, 0, stream>>>(qf, pos_u, pos_v, qub, qvb, BT * Dm);
    pe_kernel<<<(PEM * Dm) / 256, 256, 0, stream>>>(peb);
    gemm(peb, wbpos, nullptr, nullptr, 1.f, nullptr, ppb, PEM, Dm, Dm, 0);
    attn_kernel<<<Bb * Hh * (Ts / 64), 128, 0, stream>>>(qub, qvb, kb, vb, ppb, ob);
    gemm(ob, wbo, bo, x1, 1.f, x2, nullptr, BT, Dm, Dm, 0);

    // ---- conv module: x3 = x2 + pw2(silu(BN(dwconv(GLU(pw1(ln(x2))))))) ----
    ln_kernel<<<BT, 256, 0, stream>>>(x2, cv_lng, cv_lnb, nullptr, lnb);
    gemm(lnb, wbpw1, pw1_b, nullptr, 1.f, pw1f, nullptr, BT, 2 * Dm, Dm, 0);
    glu_kernel<<<(BT * Dm + 255) / 256, 256, 0, stream>>>(pw1f, gluf, BT * Dm);
    dwconv_kernel<<<(BT * Dm + 255) / 256, 256, 0, stream>>>(gluf, dw_w, dw_b, convf);
    bn_stats_kernel<<<Dm, 256, 0, stream>>>(convf, stats);
    bn_apply_kernel<<<(BT * Dm + 255) / 256, 256, 0, stream>>>(convf, stats, bn_g, bn_b, bnb, BT * Dm);
    gemm(bnb, wbpw2, pw2_b, x2, 1.f, x3, nullptr, BT, Dm, Dm, 0);

    // ---- macaron FFN #2 ----
    ln_kernel<<<BT, 256, 0, stream>>>(x3, f2_lng, f2_lnb, nullptr, lnb);
    gemm(lnb, wbf2w1, f2_b1, nullptr, 1.f, nullptr, h1b, BT, DFFc, Dm, 1);
    gemm(h1b, wbf2w2, f2_b2, x3, 0.5f, x4, nullptr, BT, Dm, DFFc, 0);

    // ---- final LayerNorm -> d_out (f32) ----
    ln_kernel<<<BT, 256, 0, stream>>>(x4, out_lng, out_lnb, (float*)d_out, nullptr);
}
